// Attention_32091995635848
// MI455X (gfx1250) — matrix operand order repaired and checked by reading, not compiled, not run on hardware
//
#include <hip/hip_runtime.h>
#include <stdint.h>

// ---------------- problem constants ----------------
constexpr int BATCH = 4;
constexpr int SEQ   = 2048;
constexpr int DIM   = 1024;           // D (input dim)
constexpr int HID   = 1024;           // H (hidden dim)
constexpr int MROWS = BATCH * SEQ;    // 8192 token rows

// ---------------- WMMA types ----------------
typedef __attribute__((ext_vector_type(16))) __bf16 v16bf;
typedef __attribute__((ext_vector_type(8)))  __bf16 v8bf;
typedef __attribute__((ext_vector_type(8)))  float  v8f;

union FragB16 { v16bf v; v8bf h[2]; };

__device__ __forceinline__ __bf16 f2bf(float f) {
  union { float f; uint32_t u; } a; a.f = f;
  uint32_t r = a.u + 0x7fffu + ((a.u >> 16) & 1u);   // round-to-nearest-even
  union { uint16_t s; __bf16 b; } o; o.s = (uint16_t)(r >> 16);
  return o.b;
}

__device__ __forceinline__ v8f vzero8() {
  v8f z;
#pragma unroll
  for (int i = 0; i < 8; ++i) z[i] = 0.0f;
  return z;
}

__device__ __forceinline__ v8f wmma_bf16(v16bf a, v16bf b, v8f c) {
  // D = A(16x32 bf16) x B(32x16 bf16) + C(16x16 f32)
  return __builtin_amdgcn_wmma_f32_16x16x32_bf16(
      /*neg_a=*/false, a, /*neg_b=*/false, b,
      /*c_mod=*/(short)0, c, /*reuse_a=*/false, /*reuse_b=*/false);
}

// ---------------- async global->LDS copy (gfx1250 ASYNCcnt path) ----------------
#if defined(__has_builtin)
#if __has_builtin(__builtin_amdgcn_global_load_async_to_lds_b128) && \
    __has_builtin(__builtin_amdgcn_s_wait_asynccnt)
#define USE_ASYNC_LDS 1
#endif
#endif
#ifndef USE_ASYNC_LDS
#define USE_ASYNC_LDS 0
#endif

// builtin parameter type: pointer to 16B int vector, in AS(1)/AS(3)
typedef int vec4i __attribute__((__vector_size__(16)));
typedef __attribute__((address_space(1))) vec4i* gas_p;
typedef __attribute__((address_space(3))) vec4i* las_p;

__device__ __forceinline__ void async_copy16(void* lds_dst, const void* gsrc) {
#if USE_ASYNC_LDS
  __builtin_amdgcn_global_load_async_to_lds_b128(
      (gas_p)(gsrc), (las_p)(lds_dst), 0, 0);
#else
  *(uint4*)lds_dst = *(const uint4*)gsrc;
#endif
}

__device__ __forceinline__ void async_wait_all() {
#if USE_ASYNC_LDS
  __builtin_amdgcn_s_wait_asynccnt(0);
#endif
}

// ---------------- kernel 1: f32 -> bf16 cast (vectorized) ----------------
__global__ __launch_bounds__(256)
void cast_f32_bf16(const float* __restrict__ in, __bf16* __restrict__ out, int n4) {
  int i = blockIdx.x * blockDim.x + threadIdx.x;
  if (i < n4) {
    float4 f = ((const float4*)in)[i];
    union { __bf16 b[4]; uint2 u; } o;
    o.b[0] = f2bf(f.x); o.b[1] = f2bf(f.y); o.b[2] = f2bf(f.z); o.b[3] = f2bf(f.w);
    ((uint2*)out)[i] = o.u;
  }
}

// ---------------- kernel 2: transpose + cast: W[R,C] f32 -> WT[C,R] bf16 ----------------
__global__ __launch_bounds__(256)
void transpose_cast(const float* __restrict__ in, __bf16* __restrict__ out, int R, int C) {
  __shared__ float tile[32][33];
  const int bx = blockIdx.x * 32;
  const int by = blockIdx.y * 32;
  const int tx = threadIdx.x & 31;
  const int ty = threadIdx.x >> 5;  // 0..7
#pragma unroll
  for (int j = ty; j < 32; j += 8)
    tile[j][tx] = in[(size_t)(by + j) * C + bx + tx];
  __syncthreads();
#pragma unroll
  for (int j = ty; j < 32; j += 8)
    out[(size_t)(bx + j) * R + by + tx] = f2bf(tile[tx][j]);
}

// ---------------- kernel 3: bf16 WMMA GEMM (256x128 block, double-buffered LDS) ---------
// C[M,N] = A[M,K] @ BT[N,K]^T + bias[N]
// 8 waves as 4(m) x 2(n); wave tile 64x64 = 4x4 WMMA accumulators.
// MODE 0: bf16 row-major; MODE 1: bf16 transposed Ct[n*M+m]; MODE 2: f32 row-major.
template <int MODE>
__global__ __launch_bounds__(256)
void gemm_bf16(const __bf16* __restrict__ A, const __bf16* __restrict__ BT,
               const float* __restrict__ bias, void* __restrict__ Cout,
               int Mdim, int Ndim, int Kdim) {
  __shared__ alignas(16) __bf16 As[2][256 * 32];   // 2 x 16KB
  __shared__ alignas(16) __bf16 Bs[2][128 * 32];   // 2 x  8KB

  const int t    = threadIdx.x;
  const int lane = t & 31;
  const int w    = t >> 5;
  const int wm   = w >> 1;        // 0..3  (64 m-rows each)
  const int wn   = w & 1;         // 0..1  (64 n-cols each)
  const int half = lane >> 4;
  const int l16  = lane & 15;
  const int m0   = blockIdx.y * 256;
  const int n0   = blockIdx.x * 128;

  v8f acc[4][4];
#pragma unroll
  for (int i = 0; i < 4; ++i)
#pragma unroll
    for (int j = 0; j < 4; ++j) acc[i][j] = vzero8();

  // stage tile(kb) into buffer buf
  auto stage = [&](int kb, int buf) {
#pragma unroll
    for (int rep = 0; rep < 4; ++rep) {            // A: 256x32 bf16 = 1024 uint4
      int i = t + rep * 256;
      int row = i >> 2, cu = i & 3;
      async_copy16((uint4*)As[buf] + i,
                   (const uint4*)(A + (size_t)(m0 + row) * Kdim + kb + cu * 8));
    }
#pragma unroll
    for (int rep = 0; rep < 2; ++rep) {            // B: 128x32 bf16 = 512 uint4
      int i = t + rep * 256;
      int row = i >> 2, cu = i & 3;
      async_copy16((uint4*)Bs[buf] + i,
                   (const uint4*)(BT + (size_t)(n0 + row) * Kdim + kb + cu * 8));
    }
  };

  stage(0, 0);
  async_wait_all();
  __syncthreads();

  int cur = 0;
  for (int kb = 0; kb < Kdim; kb += 32) {
    if (kb + 32 < Kdim) stage(kb + 32, cur ^ 1);   // fill idle buffer (async)

    // fragments from live buffer (ISA 16-bit A 16x32 / B 32x16 layouts)
    FragB16 af[4], bfr[4];
#pragma unroll
    for (int im = 0; im < 4; ++im) {
      const __bf16* p = As[cur] + (size_t)(wm * 64 + im * 16 + l16) * 32;
      af[im].h[0] = *(const v8bf*)(p + half * 8);
      af[im].h[1] = *(const v8bf*)(p + 16 + half * 8);
    }
#pragma unroll
    for (int in_ = 0; in_ < 4; ++in_) {
      const __bf16* p = Bs[cur] + (size_t)(wn * 64 + in_ * 16 + l16) * 32;
      bfr[in_].h[0] = *(const v8bf*)(p + half * 8);
      bfr[in_].h[1] = *(const v8bf*)(p + 16 + half * 8);
    }
#pragma unroll
    for (int im = 0; im < 4; ++im)
#pragma unroll
      for (int in_ = 0; in_ < 4; ++in_)
        acc[im][in_] = wmma_bf16(af[im].v, bfr[in_].v, acc[im][in_]);

    async_wait_all();     // own wave's async fills complete
    __syncthreads();      // publish next buffer / retire reads of current
    cur ^= 1;
  }

  // epilogue: C layout row = half*8+g, col = l16
#pragma unroll
  for (int im = 0; im < 4; ++im)
#pragma unroll
    for (int in_ = 0; in_ < 4; ++in_) {
      const int col = n0 + wn * 64 + in_ * 16 + l16;
      const float bv = bias[col];
#pragma unroll
      for (int g = 0; g < 8; ++g) {
        const int row = m0 + wm * 64 + im * 16 + half * 8 + g;
        const float val = acc[im][in_][g] + bv;
        if (MODE == 0)      ((__bf16*)Cout)[(size_t)row * Ndim + col] = f2bf(val);
        else if (MODE == 1) ((__bf16*)Cout)[(size_t)col * Mdim + row] = f2bf(val);
        else                ((float*)Cout)[(size_t)row * Ndim + col] = val;
      }
    }
}

// ---------------- kernel 4: flash attention (online softmax) ----------------
// Q,K: bf16 [B*S, H] row-major;  Vt: bf16 [H, B*S];  Y: bf16 [B*S, H]
// One workgroup: one batch b, 32 query rows.  8 waves:
//   scores: wave w -> 16q (w&1) x 16k (w>>1) tile over 64 keys/iter, full 1024-dot
//   P*V   : wave w -> all 32 q rows x 128-dim slice (w*128)
// K and Vt (16 MB each) stay resident in MI455X's 192 MB L2 across workgroups.
__global__ __launch_bounds__(256)
void attn_flash(const __bf16* __restrict__ Q, const __bf16* __restrict__ Kg,
                const __bf16* __restrict__ Vt, __bf16* __restrict__ Y) {
  extern __shared__ __attribute__((aligned(16))) char smem[];
  __bf16* Qs   = (__bf16*)smem;                            // 32*1024 bf16 = 64KB
  float*  Ss   = (float*)(smem + 32 * 1024 * 2);           // 32*64 f32  = 8KB
  __bf16* Ps   = (__bf16*)(smem + 32 * 1024 * 2 + 8192);   // 32*64 bf16 = 4KB
  float*  Mrow = (float*)(smem + 32 * 1024 * 2 + 8192 + 4096);
  float*  Lrow = Mrow + 32;
  float*  Scl  = Lrow + 32;
  float*  Red  = Scl + 32;                                  // 32*8 f32 reduction scratch

  const int t    = threadIdx.x;
  const int lane = t & 31;
  const int w    = t >> 5;
  const int half = lane >> 4;
  const int l16  = lane & 15;
  const int b    = blockIdx.x >> 6;            // SEQ/32 = 64 q-blocks/batch
  const int q0   = (blockIdx.x & 63) * 32;

  // stage 32x1024 Q tile (64KB, async)
  {
    const uint4* src = (const uint4*)(Q + ((size_t)b * SEQ + q0) * HID);
    for (int i = t; i < 32 * 1024 / 8; i += 256)
      async_copy16((uint4*)Qs + i, src + i);
    async_wait_all();
  }
  if (t < 32) { Mrow[t] = -__builtin_inff(); Lrow[t] = 0.0f; }
  __syncthreads();

  v8f o[2][8];
#pragma unroll
  for (int q = 0; q < 2; ++q)
#pragma unroll
    for (int d = 0; d < 8; ++d) o[q][d] = vzero8();

  const int qs    = w & 1;      // score q-subtile
  const int ks    = w >> 1;     // score k-subtile (0..3)
  const int dbase = w * 128;    // PV dim slice
  const int rr    = t >> 3;     // softmax: row (0..31)
  const int rj    = t & 7;      // softmax: 8-col segment (0..7)

  for (int kb = 0; kb < SEQ; kb += 64) {
    // ---- scores: S[16x16] = Q[16x1024] . K[16x1024]^T ----
    v8f s = vzero8();
    const __bf16* qrow = Qs + (size_t)(qs * 16 + l16) * 1024;
    const __bf16* krow = Kg + ((size_t)b * SEQ + kb + ks * 16 + l16) * HID;
    if (kb + 64 < SEQ)
      __builtin_prefetch(Kg + ((size_t)b * SEQ + kb + 64 + ks * 16 + l16) * HID, 0, 0);
#pragma unroll 4
    for (int kk = 0; kk < 1024; kk += 32) {
      FragB16 a, bb;
      a.h[0]  = *(const v8bf*)(qrow + kk + half * 8);
      a.h[1]  = *(const v8bf*)(qrow + kk + 16 + half * 8);
      bb.h[0] = *(const v8bf*)(krow + kk + half * 8);
      bb.h[1] = *(const v8bf*)(krow + kk + 16 + half * 8);
      s = wmma_bf16(a.v, bb.v, s);
    }
#pragma unroll
    for (int g = 0; g < 8; ++g)
      Ss[(size_t)(qs * 16 + half * 8 + g) * 64 + ks * 16 + l16] = s[g];
    __syncthreads();

    // ---- online softmax, 8 threads per row ----
    {
      const float* sr = Ss + (size_t)rr * 64 + rj * 8;
      float mx = -__builtin_inff();
#pragma unroll
      for (int i = 0; i < 8; ++i) mx = fmaxf(mx, sr[i]);
      Red[rr * 8 + rj] = mx;
    }
    __syncthreads();
    if (t < 32) {
      float mo = Mrow[t], mn = mo;
#pragma unroll
      for (int i = 0; i < 8; ++i) mn = fmaxf(mn, Red[t * 8 + i]);
      Mrow[t] = mn;
      Scl[t]  = __expf(mo - mn);
    }
    __syncthreads();
    {
      const float mn = Mrow[rr];
      const float* sr = Ss + (size_t)rr * 64 + rj * 8;
      __bf16* pr = Ps + (size_t)rr * 64 + rj * 8;
      float ls = 0.0f;
#pragma unroll
      for (int i = 0; i < 8; ++i) {
        float p = __expf(sr[i] - mn);
        pr[i] = f2bf(p);
        ls += p;
      }
      Red[rr * 8 + rj] = ls;
    }
    __syncthreads();
    if (t < 32) {
      float ls = 0.0f;
#pragma unroll
      for (int i = 0; i < 8; ++i) ls += Red[t * 8 + i];
      Lrow[t] = Lrow[t] * Scl[t] + ls;
    }

    // ---- rescale running O, then O += P(32x64) @ V(64 x 128-slice) ----
#pragma unroll
    for (int qq = 0; qq < 2; ++qq)
#pragma unroll
      for (int g = 0; g < 8; ++g) {
        const float sc = Scl[qq * 16 + half * 8 + g];
#pragma unroll
        for (int dt = 0; dt < 8; ++dt) o[qq][dt][g] *= sc;
      }
#pragma unroll
    for (int kc = 0; kc < 2; ++kc) {
      FragB16 ap[2];
#pragma unroll
      for (int qq = 0; qq < 2; ++qq) {
        const __bf16* pr = Ps + (size_t)(qq * 16 + l16) * 64 + kc * 32;
        ap[qq].h[0] = *(const v8bf*)(pr + half * 8);
        ap[qq].h[1] = *(const v8bf*)(pr + 16 + half * 8);
      }
      // batch V-fragment loads 4 dim-tiles at a time so 8 global_load_b128
      // are in flight before the WMMAs consume them (avoids load->wait->wmma
      // serialization seen with single-fragment register reuse)
#pragma unroll
      for (int dtc = 0; dtc < 2; ++dtc) {
        FragB16 bv4[4];
#pragma unroll
        for (int j = 0; j < 4; ++j) {
          const int dt = dtc * 4 + j;
          const __bf16* vp = Vt + (size_t)(dbase + dt * 16 + l16) * (size_t)(BATCH * SEQ)
                                + (size_t)b * SEQ + kb + kc * 32;
          if (kc == 0 && kb + 64 < SEQ) __builtin_prefetch(vp + 64, 0, 0);
          bv4[j].h[0] = *(const v8bf*)(vp + half * 8);
          bv4[j].h[1] = *(const v8bf*)(vp + 16 + half * 8);
        }
#pragma unroll
        for (int j = 0; j < 4; ++j)
#pragma unroll
          for (int qq = 0; qq < 2; ++qq)
            o[qq][dtc * 4 + j] = wmma_bf16(ap[qq].v, bv4[j].v, o[qq][dtc * 4 + j]);
      }
    }
    __syncthreads();   // protect Ps/Scl/Red before next slab
  }

  // ---- finalize: Y = O / l ----
#pragma unroll
  for (int qq = 0; qq < 2; ++qq)
#pragma unroll
    for (int g = 0; g < 8; ++g) {
      const int row = qq * 16 + half * 8 + g;
      const float inv = 1.0f / Lrow[row];
      __bf16* yr = Y + ((size_t)b * SEQ + q0 + row) * HID + dbase;
#pragma unroll
      for (int dt = 0; dt < 8; ++dt)
        yr[dt * 16 + l16] = f2bf(o[qq][dt][g] * inv);
    }
}

// ---------------- host launcher ----------------
extern "C" void kernel_launch(void* const* d_in, const int* in_sizes, int n_in,
                              void* d_out, int out_size, void* d_ws, size_t ws_size,
                              hipStream_t stream) {
  const float* x  = (const float*)d_in[0];
  const float* Wq = (const float*)d_in[1];
  const float* bq = (const float*)d_in[2];
  const float* Wk = (const float*)d_in[3];
  const float* bk = (const float*)d_in[4];
  const float* Wv = (const float*)d_in[5];
  const float* bv = (const float*)d_in[6];
  const float* Wo = (const float*)d_in[7];
  const float* bo = (const float*)d_in[8];

  // workspace partition (88 MB total)
  char* ws = (char*)d_ws;
  const size_t MB = 1ull << 20;
  __bf16* xh  = (__bf16*)(ws +  0 * MB);   // [8192,1024] bf16  (16 MB)
  __bf16* wqT = (__bf16*)(ws + 16 * MB);   // [H,D] bf16        ( 2 MB)
  __bf16* wkT = (__bf16*)(ws + 18 * MB);
  __bf16* wvT = (__bf16*)(ws + 20 * MB);
  __bf16* woT = (__bf16*)(ws + 22 * MB);   // [D,H] bf16
  __bf16* Qb  = (__bf16*)(ws + 24 * MB);   // [8192,1024] bf16  (16 MB)
  __bf16* Kb  = (__bf16*)(ws + 40 * MB);
  __bf16* Vt  = (__bf16*)(ws + 56 * MB);   // [1024,8192] bf16 (V transposed)
  __bf16* Yb  = (__bf16*)(ws + 72 * MB);

  // 1) cast x -> bf16
  cast_f32_bf16<<<(MROWS * DIM / 4 + 255) / 256, 256, 0, stream>>>(x, xh, MROWS * DIM / 4);

  // 2) transpose+cast weights: W[D,H] -> WT[H,D]; Wo[H,D] -> WoT[D,H]
  transpose_cast<<<dim3(HID / 32, DIM / 32), 256, 0, stream>>>(Wq, wqT, DIM, HID);
  transpose_cast<<<dim3(HID / 32, DIM / 32), 256, 0, stream>>>(Wk, wkT, DIM, HID);
  transpose_cast<<<dim3(HID / 32, DIM / 32), 256, 0, stream>>>(Wv, wvT, DIM, HID);
  transpose_cast<<<dim3(DIM / 32, HID / 32), 256, 0, stream>>>(Wo, woT, HID, DIM);

  // 3) QKV projections (WMMA GEMMs); V is stored transposed for the PV stage
  dim3 ggrid(HID / 128, MROWS / 256);
  gemm_bf16<0><<<ggrid, 256, 0, stream>>>(xh, wqT, bq, (void*)Qb, MROWS, HID, DIM);
  gemm_bf16<0><<<ggrid, 256, 0, stream>>>(xh, wkT, bk, (void*)Kb, MROWS, HID, DIM);
  gemm_bf16<1><<<ggrid, 256, 0, stream>>>(xh, wvT, bv, (void*)Vt, MROWS, HID, DIM);

  // 4) flash attention
  const size_t smem = 32 * 1024 * 2 /*Qs*/ + 8192 /*Ss*/ + 4096 /*Ps*/
                    + 3 * 32 * 4 /*stats*/ + 32 * 8 * 4 /*Red*/;
  attn_flash<<<BATCH * (SEQ / 32), 256, smem, stream>>>(Qb, Kb, Vt, Yb);

  // 5) output projection -> f32 d_out
  gemm_bf16<2><<<dim3(DIM / 128, MROWS / 256), 256, 0, stream>>>(Yb, woT, bo, d_out, MROWS, DIM, HID);
}
